// MetaQDA_MAP_4097398800824
// MI455X (gfx1250) — matrix operand order repaired and checked by reading, compiled, not run on hardware
//
#include <hip/hip_runtime.h>
#include <hip/hip_bf16.h>

// MetaQDA MAP on MI455X (gfx1250, wave32).
// Every heavy op is cast as C = A * B^T with row-major operands, so both WMMA
// fragments are two contiguous 16B global loads per lane (L2-resident data,
// 192MB L2 >> ~110MB working set). No LDS in the GEMM inner loops.
// Matrix inversion = GEMM-only Newton-Schulz (runs entirely on v_wmma).

typedef __bf16 bf16;
typedef __attribute__((ext_vector_type(16))) __bf16 v16bf;
typedef __attribute__((ext_vector_type(8)))  __bf16 v8bf;
typedef __attribute__((ext_vector_type(8)))  float  v8f;

#define Nn 4096
#define Mm 4096
#define Dd 1024
#define Cc 16
#define NITER 24

// ---------------------------------------------------------------- WMMA core
__device__ __forceinline__ v8f wmma_bf(v16bf a, v16bf b, v8f c) {
  return __builtin_amdgcn_wmma_f32_16x16x32_bf16(false, a, false, b,
                                                 (short)0, c, false, false);
}

__device__ __forceinline__ v16bf cat8(v8bf lo, v8bf hi) {
  return __builtin_shufflevector(lo, hi, 0, 1, 2, 3, 4, 5, 6, 7,
                                 8, 9, 10, 11, 12, 13, 14, 15);
}

// A fragment (16x32 MxK): lane(m=l&15,h=l>>4) holds K = {8h..8h+7, 16+8h..23+8h}.
__device__ __forceinline__ v16bf load_a(const bf16* __restrict__ row, int k0, int h) {
  v8bf lo = *(const v8bf*)(row + k0 + 8 * h);
  v8bf hi = *(const v8bf*)(row + k0 + 16 + 8 * h);
  return cat8(lo, hi);
}
// Same, with a bf16 mask applied along K (mask index == k, fully data-parallel).
__device__ __forceinline__ v16bf load_a_masked(const bf16* __restrict__ row,
                                               const bf16* __restrict__ mrow,
                                               int k0, int h) {
  v8bf lo = *(const v8bf*)(row + k0 + 8 * h);
  v8bf hi = *(const v8bf*)(row + k0 + 16 + 8 * h);
  v8bf ml = *(const v8bf*)(mrow + k0 + 8 * h);
  v8bf mh = *(const v8bf*)(mrow + k0 + 16 + 8 * h);
  v8bf olo, ohi;
  #pragma unroll
  for (int e = 0; e < 8; ++e) {
    olo[e] = (bf16)((float)lo[e] * (float)ml[e]);
    ohi[e] = (bf16)((float)hi[e] * (float)mh[e]);
  }
  return cat8(olo, ohi);
}
// B^T fragment (32x16 KxN from row-major B[N][K]): lane(n=l&15,h) holds row n,
// K = {8h..8h+7, 16+8h..23+8h} (same order as A).
__device__ __forceinline__ v16bf load_bt(const bf16* __restrict__ row, int k0, int h) {
  v8bf lo = *(const v8bf*)(row + k0 + 8 * h);
  v8bf hi = *(const v8bf*)(row + k0 + 16 + 8 * h);
  return cat8(lo, hi);
}

// ---------------------------------------------------------------- small utils
__global__ void zero_f32(float* p, int n) {
  int i = blockIdx.x * 256 + threadIdx.x;
  if (i < n) p[i] = 0.f;
}

__global__ void f32_to_bf16(const float* __restrict__ s, bf16* __restrict__ d, int n) {
  int i = blockIdx.x * 256 + threadIdx.x;
  if (i < n) d[i] = (bf16)s[i];
}

// XT[j][k] = bf16(X[k][j])  (one-time 8MB transpose; L2-resident)
__global__ void xt_kernel(const float* __restrict__ X, bf16* __restrict__ XT) {
  int idx = blockIdx.x * 256 + threadIdx.x;   // Dd*Nn
  int j = idx >> 12, k = idx & (Nn - 1);
  XT[idx] = (bf16)X[(size_t)k * Dd + j];
}

// maskbf[c][n] = (y[n]==c) ? 1 : 0  (bf16, 128KB table)
__global__ void mask_kernel(const int* __restrict__ y, bf16* __restrict__ mk) {
  int idx = blockIdx.x * 256 + threadIdx.x;   // Cc*Nn
  int c = idx >> 12, n = idx & (Nn - 1);
  mk[idx] = (bf16)((y[n] == c) ? 1.f : 0.f);
}

__global__ void build_Lbf(const float* __restrict__ tdiag,
                          const float* __restrict__ tlow, bf16* __restrict__ Lbf) {
  int idx = blockIdx.x * 256 + threadIdx.x;   // Dd*Dd
  int i = idx >> 10, j = idx & (Dd - 1);
  float v = (i > j) ? tlow[idx] : (i == j ? fabsf(tdiag[i]) : 0.f);
  Lbf[idx] = (bf16)v;
}

__global__ void nj_kernel(const int* __restrict__ y, float* __restrict__ Nj) {
  __shared__ int cnt[Cc];
  int t = threadIdx.x;
  if (t < Cc) cnt[t] = 0;
  __syncthreads();
  for (int n = t; n < Nn; n += 256) atomicAdd(&cnt[y[n]], 1);
  __syncthreads();
  if (t < Cc) Nj[t] = (float)cnt[t];
}

__global__ void stats_kernel(const float* __restrict__ X, const int* __restrict__ y,
                             float* __restrict__ sums) {
  int j = blockIdx.x * 256 + threadIdx.x;     // Dd threads
  float acc[Cc];
  #pragma unroll
  for (int c = 0; c < Cc; ++c) acc[c] = 0.f;
  for (int n = 0; n < Nn; ++n) {
    float x = X[(size_t)n * Dd + j];
    int cv = y[n];
    #pragma unroll
    for (int c = 0; c < Cc; ++c) acc[c] += (cv == c) ? x : 0.f;
  }
  #pragma unroll
  for (int c = 0; c < Cc; ++c) sums[c * Dd + j] = acc[c];
}

__global__ void prep_kernel(const float* __restrict__ mvec, const float* __restrict__ kappa,
                            const float* __restrict__ nu, const float* __restrict__ Nj,
                            const float* __restrict__ sums, float* __restrict__ mu,
                            float* __restrict__ kapNj, float* __restrict__ invden) {
  int idx = blockIdx.x * 256 + threadIdx.x;   // Cc*Dd
  int c = idx >> 10, j = idx & (Dd - 1);
  float kap = fabsf(kappa[0]) + 1e-6f;
  float nj = Nj[c];
  mu[idx] = (kap * mvec[j] + sums[idx]) / (kap + nj);
  if (j == 0) {
    kapNj[c] = kap + nj;
    float nup = fmaxf(nu[0], (float)Dd - 1.f + 1e-6f);
    invden[c] = 1.f / (nup + nj + (float)Dd + 2.f);
  }
}

// ------------------------------------------------- sigma assembly
// sigma[c] = ((W_c X)^T X + L L^T + kap m m^T - (kap+Nj) mu mu^T) / den
// Pass 1: gemmNT(XT masked-along-K, XT) over K=Nn.  Pass 2: gemmNT(L, L) over K=Dd.
// Tile: 16 (M) x 64 (N) per wave, 4 accumulators sharing one A fragment.
__global__ void __launch_bounds__(32)
gram_sigma_kernel(const bf16* __restrict__ XT, const bf16* __restrict__ maskbf,
                  const bf16* __restrict__ Lbf, const float* __restrict__ mvec,
                  const float* __restrict__ kappa, const float* __restrict__ mu,
                  const float* __restrict__ kapNj, const float* __restrict__ invden,
                  bf16* __restrict__ Sigbf, float* __restrict__ trace) {
  const int lane = threadIdx.x;
  const int c = blockIdx.y;
  const int TJ = Dd / 64;                     // 16
  const int ti = blockIdx.x / TJ, tj = blockIdx.x % TJ;
  const int i0 = ti * 16, j0 = tj * 64;
  const int m = lane & 15, h = lane >> 4;

  v8f acc[4];
  #pragma unroll
  for (int nb = 0; nb < 4; ++nb) acc[nb] = (v8f){0.f,0.f,0.f,0.f,0.f,0.f,0.f,0.f};

  // Pass 1: masked Gram, K = samples
  {
    const bf16* arow = XT + (size_t)(i0 + m) * Nn;
    const bf16* mrow = maskbf + (size_t)c * Nn;
    const bf16* b0 = XT + (size_t)(j0 + m) * Nn;
    for (int k0 = 0; k0 < Nn; k0 += 32) {
      __builtin_prefetch(arow + k0 + 256, 0, 0);
      v16bf a = load_a_masked(arow, mrow, k0, h);
      #pragma unroll
      for (int nb = 0; nb < 4; ++nb)
        acc[nb] = wmma_bf(a, load_bt(b0 + (size_t)nb * 16 * Nn, k0, h), acc[nb]);
    }
  }
  // Pass 2: prior L*L^T, K = features
  {
    const bf16* arow = Lbf + (size_t)(i0 + m) * Dd;
    const bf16* b0 = Lbf + (size_t)(j0 + m) * Dd;
    for (int k0 = 0; k0 < Dd; k0 += 32) {
      v16bf a = load_a(arow, k0, h);
      #pragma unroll
      for (int nb = 0; nb < 4; ++nb)
        acc[nb] = wmma_bf(a, load_bt(b0 + (size_t)nb * 16 * Dd, k0, h), acc[nb]);
    }
  }
  // Epilogue: rank-1 terms, scale, store, trace.
  const float kap = fabsf(kappa[0]) + 1e-6f;
  const float kn = kapNj[c], inv = invden[c];
  const int mb = 8 * h;
  #pragma unroll
  for (int nb = 0; nb < 4; ++nb) {
    const int gj = j0 + nb * 16 + m;
    const float mjv = mvec[gj];
    const float muj = mu[c * Dd + gj];
    #pragma unroll
    for (int r = 0; r < 8; ++r) {
      const int gi = i0 + mb + r;
      float v = (acc[nb][r] + kap * mvec[gi] * mjv - kn * mu[c * Dd + gi] * muj) * inv;
      Sigbf[((size_t)c * Dd + gi) * Dd + gj] = (bf16)v;
      if (gi == gj) atomicAdd(&trace[c], v);
    }
  }
}

// ------------------------------------------------- Newton-Schulz kernels
__global__ void init_Y(float* __restrict__ Yc, const float* __restrict__ trace, int c) {
  int idx = blockIdx.x * 256 + threadIdx.x;   // Dd*Dd
  int i = idx >> 10, j = idx & (Dd - 1);
  Yc[idx] = (i == j) ? (1.0f / trace[c]) : 0.0f;
}

// Tt = (A * B^T)^T  : A = Sigma (sym), B = Ybf (sym)  =>  Tt = (Sigma*Y)^T
__global__ void __launch_bounds__(32)
gemm_nt_t_kernel(const bf16* __restrict__ A, const bf16* __restrict__ B,
                 bf16* __restrict__ CoT) {
  const int lane = threadIdx.x;
  const int TJ = Dd / 64;
  const int ti = blockIdx.x / TJ, tj = blockIdx.x % TJ;
  const int i0 = ti * 16, j0 = tj * 64;
  const int m = lane & 15, h = lane >> 4;
  v8f acc[4];
  #pragma unroll
  for (int nb = 0; nb < 4; ++nb) acc[nb] = (v8f){0.f,0.f,0.f,0.f,0.f,0.f,0.f,0.f};
  const bf16* arow = A + (size_t)(i0 + m) * Dd;
  const bf16* b0 = B + (size_t)(j0 + m) * Dd;
  for (int k0 = 0; k0 < Dd; k0 += 32) {
    __builtin_prefetch(arow + k0 + 256, 0, 0);
    v16bf a = load_a(arow, k0, h);
    #pragma unroll
    for (int nb = 0; nb < 4; ++nb)
      acc[nb] = wmma_bf(a, load_bt(b0 + (size_t)nb * 16 * Dd, k0, h), acc[nb]);
  }
  const int mb = 8 * h;
  #pragma unroll
  for (int nb = 0; nb < 4; ++nb) {
    const int gj = j0 + nb * 16 + m;
    #pragma unroll
    for (int r = 0; r < 8; ++r)
      CoT[(size_t)gj * Dd + (i0 + mb + r)] = (bf16)acc[nb][r];   // transposed store
  }
}

// Y = 2*Y - A * B^T  : A = Ybf, B = Tt  =>  Y = 2Y - Y*(Sigma*Y)
__global__ void __launch_bounds__(32)
gemm_nt_newton_kernel(const bf16* __restrict__ A, const bf16* __restrict__ B,
                      float* __restrict__ Y) {
  const int lane = threadIdx.x;
  const int TJ = Dd / 64;
  const int ti = blockIdx.x / TJ, tj = blockIdx.x % TJ;
  const int i0 = ti * 16, j0 = tj * 64;
  const int m = lane & 15, h = lane >> 4;
  v8f acc[4];
  #pragma unroll
  for (int nb = 0; nb < 4; ++nb) acc[nb] = (v8f){0.f,0.f,0.f,0.f,0.f,0.f,0.f,0.f};
  const bf16* arow = A + (size_t)(i0 + m) * Dd;
  const bf16* b0 = B + (size_t)(j0 + m) * Dd;
  for (int k0 = 0; k0 < Dd; k0 += 32) {
    __builtin_prefetch(arow + k0 + 256, 0, 0);
    v16bf a = load_a(arow, k0, h);
    #pragma unroll
    for (int nb = 0; nb < 4; ++nb)
      acc[nb] = wmma_bf(a, load_bt(b0 + (size_t)nb * 16 * Dd, k0, h), acc[nb]);
  }
  const int mb = 8 * h;
  #pragma unroll
  for (int nb = 0; nb < 4; ++nb) {
    const int gj = j0 + nb * 16 + m;
    #pragma unroll
    for (int r = 0; r < 8; ++r) {
      const size_t idx = (size_t)(i0 + mb + r) * Dd + gj;
      Y[idx] = 2.0f * Y[idx] - acc[nb][r];
    }
  }
}

// Sigbf[c] <- bf16(0.9*Y + 0.1*I)
__global__ void blend_kernel(const float* __restrict__ Yc, bf16* __restrict__ Sigc) {
  int idx = blockIdx.x * 256 + threadIdx.x;   // Dd*Dd
  int i = idx >> 10, j = idx & (Dd - 1);
  Sigc[idx] = (bf16)(0.9f * Yc[idx] + ((i == j) ? 0.1f : 0.f));
}

// ------------------------------------------------- predict: fused quadratic form
// logits[m][c] = -sum_j (diff * Sinv)[m][j] * diff[m][j];  Sinv symmetric => NT form.
__global__ void __launch_bounds__(32)
logits_kernel(const float* __restrict__ Xq, const float* __restrict__ mu,
              const bf16* __restrict__ Sinv, float* __restrict__ out) {
  const int lane = threadIdx.x;
  const int c = blockIdx.y;
  const int TJ = Dd / 64;
  const int ti = blockIdx.x / TJ, tj = blockIdx.x % TJ;
  const int m0 = ti * 16, j0 = tj * 64;
  const int m = lane & 15, h = lane >> 4;
  v8f acc[4];
  #pragma unroll
  for (int nb = 0; nb < 4; ++nb) acc[nb] = (v8f){0.f,0.f,0.f,0.f,0.f,0.f,0.f,0.f};

  const float* xrow = Xq + (size_t)(m0 + m) * Dd;
  const float* muc = mu + c * Dd;
  const bf16* b0 = Sinv + ((size_t)c * Dd + j0 + m) * Dd;

  for (int k0 = 0; k0 < Dd; k0 += 32) {
    // A fragment: diff = Xq - mu computed in registers, converted to bf16.
    v8f xlo = *(const v8f*)(xrow + k0 + 8 * h);
    v8f xhi = *(const v8f*)(xrow + k0 + 16 + 8 * h);
    v8f mlo = *(const v8f*)(muc + k0 + 8 * h);
    v8f mhi = *(const v8f*)(muc + k0 + 16 + 8 * h);
    v8f dlo = xlo - mlo, dhi = xhi - mhi;
    v16bf a;
    #pragma unroll
    for (int e = 0; e < 8; ++e) { a[e] = (bf16)dlo[e]; a[e + 8] = (bf16)dhi[e]; }
    #pragma unroll
    for (int nb = 0; nb < 4; ++nb)
      acc[nb] = wmma_bf(a, load_bt(b0 + (size_t)nb * 16 * Dd, k0, h), acc[nb]);
  }
  // Fused row-dot epilogue: reduce over the 16 column-lanes per output row.
  const int mb = 8 * h;
  #pragma unroll
  for (int nb = 0; nb < 4; ++nb) {
    const int gj = j0 + nb * 16 + m;
    const float mju = muc[gj];
    #pragma unroll
    for (int r = 0; r < 8; ++r) {
      const int gm = m0 + mb + r;
      float p = acc[nb][r] * (Xq[(size_t)gm * Dd + gj] - mju);
      #pragma unroll
      for (int off = 8; off >= 1; off >>= 1) p += __shfl_xor(p, off, 16);
      if (m == r) atomicAdd(&out[gm * Cc + c], -p);
    }
  }
}

// ---------------------------------------------------------------- launcher
extern "C" void kernel_launch(void* const* d_in, const int* in_sizes, int n_in,
                              void* d_out, int out_size, void* d_ws, size_t ws_size,
                              hipStream_t stream) {
  const float* X     = (const float*)d_in[0];
  const int*   y     = (const int*)  d_in[1];
  const float* Xq    = (const float*)d_in[2];
  const float* mvec  = (const float*)d_in[3];
  const float* kappa = (const float*)d_in[4];
  const float* nu    = (const float*)d_in[5];
  const float* tdiag = (const float*)d_in[6];
  const float* tlow  = (const float*)d_in[7];
  float* out = (float*)d_out;

  char* ws = (char*)d_ws;
  size_t off = 0;
  auto take = [&](size_t bytes) -> char* {
    char* p = ws + off;
    off = (off + bytes + 255) & ~(size_t)255;
    return p;
  };
  bf16*  Sigbf  = (bf16*) take((size_t)Cc * Dd * Dd * sizeof(bf16));  // 32 MB
  float* Y      = (float*)take((size_t)Cc * Dd * Dd * sizeof(float)); // 64 MB
  bf16*  ybf    = (bf16*) take((size_t)Dd * Dd * sizeof(bf16));       //  2 MB
  bf16*  tbf    = (bf16*) take((size_t)Dd * Dd * sizeof(bf16));       //  2 MB
  bf16*  maskbf = (bf16*) take((size_t)Cc * Nn * sizeof(bf16));       // 128 KB
  float* sums   = (float*)take((size_t)Cc * Dd * sizeof(float));
  float* mu     = (float*)take((size_t)Cc * Dd * sizeof(float));
  float* Nj     = (float*)take(Cc * sizeof(float));
  float* kapNj  = (float*)take(Cc * sizeof(float));
  float* invden = (float*)take(Cc * sizeof(float));
  float* trace  = (float*)take(Cc * sizeof(float));
  // Transient bf16 operands alias the Y region (fully consumed by
  // gram_sigma_kernel before any init_Y writes Y — stream ordered).
  bf16* XT  = (bf16*)Y;                                               // 8 MB
  bf16* Lbf = (bf16*)((char*)Y + (size_t)Dd * Nn * sizeof(bf16));     // 2 MB

  zero_f32<<<(Mm * Cc + 255) / 256, 256, 0, stream>>>(out, Mm * Cc);
  zero_f32<<<1, 256, 0, stream>>>(trace, Cc);
  nj_kernel<<<1, 256, 0, stream>>>(y, Nj);
  stats_kernel<<<Dd / 256, 256, 0, stream>>>(X, y, sums);
  prep_kernel<<<(Cc * Dd) / 256, 256, 0, stream>>>(mvec, kappa, nu, Nj, sums,
                                                   mu, kapNj, invden);
  xt_kernel<<<(Dd * Nn) / 256, 256, 0, stream>>>(X, XT);
  mask_kernel<<<(Cc * Nn) / 256, 256, 0, stream>>>(y, maskbf);
  build_Lbf<<<(Dd * Dd) / 256, 256, 0, stream>>>(tdiag, tlow, Lbf);

  dim3 gsig((Dd / 16) * (Dd / 64), Cc);                // 1024 x 16
  gram_sigma_kernel<<<gsig, 32, 0, stream>>>(XT, maskbf, Lbf, mvec, kappa, mu,
                                             kapNj, invden, Sigbf, trace);

  const int gtiles = (Dd / 16) * (Dd / 64);            // 1024
  for (int c = 0; c < Cc; ++c) {
    float* Yc = Y + (size_t)c * Dd * Dd;
    bf16*  Sc = Sigbf + (size_t)c * Dd * Dd;
    init_Y<<<(Dd * Dd) / 256, 256, 0, stream>>>(Yc, trace, c);
    for (int it = 0; it < NITER; ++it) {
      f32_to_bf16<<<(Dd * Dd) / 256, 256, 0, stream>>>(Yc, ybf, Dd * Dd);
      gemm_nt_t_kernel<<<gtiles, 32, 0, stream>>>(Sc, ybf, tbf);      // Tt=(Sig*Y)^T
      gemm_nt_newton_kernel<<<gtiles, 32, 0, stream>>>(ybf, tbf, Yc); // Y=2Y-Y*T
    }
    blend_kernel<<<(Dd * Dd) / 256, 256, 0, stream>>>(Yc, Sc);        // Sc=0.9Y+0.1I
  }

  dim3 glog((Mm / 16) * (Dd / 64), Cc);                // 4096 x 16
  logits_kernel<<<glog, 32, 0, stream>>>(Xq, mu, Sigbf, out);
}
